// MACE_60559038874213
// MI455X (gfx1250) — matrix operand order repaired and checked by reading, not compiled, not run on hardware
//
#include <hip/hip_runtime.h>
#include <hip/hip_bf16.h>

#define DEV __device__ __forceinline__

typedef __attribute__((ext_vector_type(16))) __bf16 v16bf;
typedef __attribute__((ext_vector_type(8)))  float  v8f;

static constexpr int NN   = 10000;
static constexpr int EE   = 160000;
static constexpr int GG   = 16;
static constexpr int ZZ   = 10;
static constexpr int KK   = 64;
static constexpr int NBAS = 8;
static constexpr int HIDN = 16;
static constexpr float RMAXF   = 5.0f;
static constexpr float INV_AVG = 1.0f / 16.0f;
static constexpr float PI_F    = 3.14159265358979323846f;

DEV int laneid() { return (int)(threadIdx.x & 31u); }

DEV v8f vzero8() { v8f c;
#pragma unroll
  for (int i = 0; i < 8; ++i) c[i] = 0.f;
  return c; }

DEV float siluf(float x)  { float s = 1.f / (1.f + __expf(-x)); return x * s; }
DEV float dsiluf(float x) { float s = 1.f / (1.f + __expf(-x)); return s * (1.f + x * (1.f - s)); }

#define WMMA_BF16(A, B, C) \
  __builtin_amdgcn_wmma_f32_16x16x32_bf16(false, (A), false, (B), (short)0, (C), false, false)

// ---------- CDNA5 async global->LDS copy (ASYNCcnt path) ----------
DEV unsigned lds_off(const void* p) {            // flat LDS aperture: low 32 bits = LDS offset
  return (unsigned)(unsigned long long)p;
}
DEV void async_b128(unsigned ldsDst, const void* g) {
  asm volatile("global_load_async_to_lds_b128 %0, %1, off"
               :: "v"(ldsDst), "v"(g) : "memory");
}
DEV void wait_async0() { asm volatile("s_wait_asynccnt 0x0" ::: "memory"); }

// ---- A fragment: 16x32 bf16 tile, row-major LDS [16][ld], column base kbase.
// lanes 0-15: K in {0..7, 16..23}; lanes 16-31: K in {8..15, 24..31} (ISA 7.12.2).
DEV v16bf fragA(const __bf16* s, int ld, int kbase) {
  int l = laneid(); int m = l & 15; int half = l >> 4;
  const __bf16* p = s + m * ld + kbase + 8 * half;
  v16bf a;
#pragma unroll
  for (int t = 0; t < 8; ++t) {
    int off = ((t >> 2) << 4) + ((t & 3) << 1);
    a[2 * t]     = p[off];
    a[2 * t + 1] = p[off + 1];
  }
  return a;
}

// ---- B fragment: 32x16 bf16 tile from a column-stage: stage[n][k] = B[k][n].
// lane holds column n = lane&15; K = 8*(lane>>4) + {0..7, 16..23} (matches fragA).
DEV v16bf fragB(const __bf16* s, int ld, int kbase) {
  int l = laneid(); int n = l & 15; int half = l >> 4;
  const __bf16* p = s + n * ld + kbase + 8 * half;
  v16bf b;
#pragma unroll
  for (int t = 0; t < 16; ++t) b[t] = p[t < 8 ? t : t + 8];
  return b;
}

// one dense 16x64 = 16x64 @ 64x64 step (two K-chunks, four N-tiles)
DEV void gemm64(const __bf16* xin, const __bf16* Bst, v8f c[4]) {
  v16bf a0 = fragA(xin, 64, 0);
  v16bf a1 = fragA(xin, 64, 32);
#pragma unroll
  for (int nt = 0; nt < 4; ++nt) {
    v8f acc = vzero8();
    acc = WMMA_BF16(a0, fragB(Bst + nt * 16 * 64, 64, 0),  acc);
    acc = WMMA_BF16(a1, fragB(Bst + nt * 16 * 64, 64, 32), acc);
    c[nt] = acc;
  }
}

template <bool SAVE_PRE, bool DO_SILU>
DEV void storeAct(__bf16* act, __bf16* pre, const v8f c[4]) {
  int l = laneid(); int n = l & 15; int half = l >> 4;
#pragma unroll
  for (int nt = 0; nt < 4; ++nt)
#pragma unroll
    for (int g = 0; g < 8; ++g) {
      int m = g + 8 * half;
      float v = c[nt][g];
      if (SAVE_PRE) pre[m * 64 + nt * 16 + n] = (__bf16)v;
      act[m * 64 + nt * 16 + n] = (__bf16)(DO_SILU ? siluf(v) : v);
    }
}

DEV void multStore(__bf16* dst, const __bf16* pre, const v8f c[4]) {
  int l = laneid(); int n = l & 15; int half = l >> 4;
#pragma unroll
  for (int nt = 0; nt < 4; ++nt)
#pragma unroll
    for (int g = 0; g < 8; ++g) {
      int m = g + 8 * half;
      float pv = (float)pre[m * 64 + nt * 16 + n];
      dst[m * 64 + nt * 16 + n] = (__bf16)(c[nt][g] * dsiluf(pv));
    }
}

// =============================== weight staging ===============================
// dstT[n*ldT+k] = (k<rows)? src[k*cols+n] : 0   (column stage for forward B frags)
// dstR[r*cols+c] = (r<rows)? src[r*cols+c] : 0  (row stage for transposed/backward B frags)
__global__ void k_stage_w(const float* __restrict__ src, int rows, int cols,
                          int ldT, int rPad, __bf16* dstT, __bf16* dstR) {
  int i = blockIdx.x * blockDim.x + threadIdx.x;
  int nT = cols * ldT;
  if (i < nT) {
    int n = i / ldT, k = i % ldT;
    dstT[i] = (__bf16)((k < rows) ? src[k * cols + n] : 0.f);
  }
  if (i < rPad * cols) {
    int r = i / cols, c = i % cols;
    dstR[i] = (__bf16)((r < rows) ? src[r * cols + c] : 0.f);
  }
}

// =============================== edge geometry ================================
__global__ void k_edge_geom(const float* __restrict__ pos, const float* __restrict__ shifts,
                            const int* __restrict__ sender, const int* __restrict__ receiver,
                            float* __restrict__ u, float* __restrict__ rad,
                            float* __restrict__ drad, int nE) {
  int e = blockIdx.x * blockDim.x + threadIdx.x;
  if (e >= nE) return;
  int s = sender[e], r = receiver[e];
  float dx = pos[r * 3 + 0] - pos[s * 3 + 0] + shifts[e * 3 + 0];
  float dy = pos[r * 3 + 1] - pos[s * 3 + 1] + shifts[e * 3 + 1];
  float dz = pos[r * 3 + 2] - pos[s * 3 + 2] + shifts[e * 3 + 2];
  float rr  = sqrtf(dx * dx + dy * dy + dz * dz + 1e-9f);
  float inv = 1.f / rr;
  u[e * 3 + 0] = dx * inv; u[e * 3 + 1] = dy * inv; u[e * 3 + 2] = dz * inv;
  float x = rr / RMAXF;
  float f = 0.f, df = 0.f;
  if (x < 1.f) {  // P=5 polynomial cutoff and its r-derivative
    float x2 = x * x, x4 = x2 * x2, x5 = x4 * x, x6 = x5 * x, x7 = x6 * x;
    f  = 1.f - 21.f * x5 + 35.f * x6 - 15.f * x7;
    df = (-105.f * x4 + 210.f * x5 - 105.f * x6) / RMAXF;
  }
  const float C = sqrtf(2.f / RMAXF);
#pragma unroll
  for (int b = 0; b < NBAS; ++b) {
    float an = (float)(b + 1) * PI_F / RMAXF;
    float sn = __sinf(an * rr), cn = __cosf(an * rr);
    float bess  = C * sn * inv;
    float dbess = C * (an * cn * inv - sn * inv * inv);
    rad[e * NBAS + b]  = bess * f;
    drad[e * NBAS + b] = dbess * f + bess * df;
  }
}

// ============================ node init / embedding ===========================
__global__ void k_elem(const float* __restrict__ attrs, const float* __restrict__ aE,
                       const int* __restrict__ batch, int* __restrict__ elem,
                       float* __restrict__ outE, int nN) {
  int n = blockIdx.x * blockDim.x + threadIdx.x;
  if (n >= nN) return;
  int z = 0;
  for (int t = 0; t < ZZ; ++t) if (attrs[n * ZZ + t] > 0.5f) z = t;
  elem[n] = z;
  atomicAdd(&outE[batch[n]], aE[z]);
}

__global__ void k_embed(const float* __restrict__ W_embed, const int* __restrict__ elem,
                        float* __restrict__ feats0, int total) {
  int i = blockIdx.x * blockDim.x + threadIdx.x;
  if (i >= total) return;
  int n = i >> 6, k = i & 63;
  feats0[i] = W_embed[elem[n] * KK + k];
}

// ============================ generic wave GEMM ===============================
// Y[16-tile] = scale * (X @ Bstage) (+ Cinit when ACC; scale must be 1 then)
// nRows must be a multiple of 16 (N = 10000 = 625*16) -> no row guards.
template <int WAVES, bool ACC>
__global__ void k_gemm(const float* __restrict__ X, const __bf16* __restrict__ Bst,
                       const float* __restrict__ Cinit, float* __restrict__ Y,
                       int nRows, float scale) {
  __shared__ __align__(16) __bf16 sX[WAVES][16 * 64];
  __shared__ __align__(16) __bf16 sW[64 * 64];
  int t = (int)threadIdx.x;
  { // async-stage the 8KB B matrix into LDS (32B per thread, 256 threads)
    unsigned dst = lds_off(sW) + t * 32;
    const char* src = (const char*)Bst + t * 32;
    async_b128(dst, src);
    async_b128(dst + 16, src + 16);
    wait_async0();
  }
  __syncthreads();
  int wv = t >> 5;
  int tile = blockIdx.x * WAVES + wv;
  int nTiles = nRows >> 4;
  if (tile >= nTiles) return;
  int r0 = tile * 16;
  __bf16* xa = sX[wv];
  int l = laneid(), lc = l & 15, lh = l >> 4;
  for (int idx = l; idx < 16 * 64; idx += 32) {
    int m = idx >> 6, k = idx & 63;
    xa[idx] = (__bf16)X[(r0 + m) * 64 + k];
  }
  v16bf a0 = fragA(xa, 64, 0), a1 = fragA(xa, 64, 32);
#pragma unroll
  for (int nt = 0; nt < 4; ++nt) {
    v8f c = vzero8();
    if (ACC) {
#pragma unroll
      for (int g = 0; g < 8; ++g)
        c[g] = Cinit[(r0 + g + 8 * lh) * 64 + nt * 16 + lc];
    }
    c = WMMA_BF16(a0, fragB(sW + nt * 16 * 64, 64, 0),  c);
    c = WMMA_BF16(a1, fragB(sW + nt * 16 * 64, 64, 32), c);
#pragma unroll
    for (int g = 0; g < 8; ++g)
      Y[(r0 + g + 8 * lh) * 64 + nt * 16 + lc] = c[g] * scale;
  }
}

// ===================== edge MLP forward + message scatter =====================
// grid tiles E exactly (E = 160000 = 1250*8*16): every wave is live -> barrier safe.
template <int WAVES>
__global__ void k_edge_fwd(const float* __restrict__ rad, const float* __restrict__ h,
                           const int* __restrict__ sender, const int* __restrict__ receiver,
                           const __bf16* __restrict__ R1t, const __bf16* __restrict__ R2t,
                           const __bf16* __restrict__ R3t, const __bf16* __restrict__ R4t,
                           float* __restrict__ A0, int nEdges) {
  __shared__ __align__(16) __bf16 sA[WAVES][16 * 64];
  __shared__ __align__(16) __bf16 sB[WAVES][16 * 64];
  __shared__ __align__(16) __bf16 sW[2048 + 3 * 4096];   // R1t | R2t | R3t | R4t (28KB)
  int t = (int)threadIdx.x;
  { // async-stage all four forward weight stages into LDS
    async_b128(lds_off(sW) + t * 16, (const char*)R1t + t * 16);          // 4KB
    const __bf16* srcs[3] = {R2t, R3t, R4t};
#pragma unroll
    for (int m = 0; m < 3; ++m) {
      unsigned dst = lds_off(sW + 2048 + m * 4096) + t * 32;
      const char* src = (const char*)srcs[m] + t * 32;
      async_b128(dst, src);
      async_b128(dst + 16, src + 16);
    }
    wait_async0();
  }
  __syncthreads();
  const __bf16* wR1 = sW;
  const __bf16* wR2 = sW + 2048;
  const __bf16* wR3 = sW + 2048 + 4096;
  const __bf16* wR4 = sW + 2048 + 2 * 4096;

  int wv = t >> 5;
  int tile = blockIdx.x * WAVES + wv;
  if (tile * 16 >= nEdges) return;
  int e0 = tile * 16;
  __bf16 *xa = sA[wv], *xb = sB[wv];
  int l = laneid(), lc = l & 15, lh = l >> 4;

  for (int idx = l; idx < 16 * 32; idx += 32) {
    int m = idx >> 5, k = idx & 31;
    xa[m * 64 + k] = (__bf16)((k < NBAS) ? rad[(e0 + m) * NBAS + k] : 0.f);
  }
  v8f c[4];
  {
    v16bf a = fragA(xa, 64, 0);
#pragma unroll
    for (int nt = 0; nt < 4; ++nt) {
      v8f acc = vzero8();
      acc = WMMA_BF16(a, fragB(wR1 + nt * 16 * 32, 32, 0), acc);
      c[nt] = acc;
    }
  }
  storeAct<false, true>(xb, nullptr, c);
  gemm64(xb, wR2, c); storeAct<false, true>(xa, nullptr, c);
  gemm64(xa, wR3, c); storeAct<false, true>(xb, nullptr, c);
  gemm64(xb, wR4, c);                    // rw, no activation
  // m0 = rw * h[sender]  (spherical channel 0, sh==1); scatter to A0[receiver]
#pragma unroll
  for (int g = 0; g < 8; ++g) {
    int m = g + 8 * lh;
    int e = e0 + m;
    int s = sender[e], r = receiver[e];
#pragma unroll
    for (int nt = 0; nt < 4; ++nt) {
      int k = nt * 16 + lc;
      atomicAdd(&A0[r * 64 + k], c[nt][g] * h[s * 64 + k]);
    }
  }
}

// ================= edge backward: recompute MLP, transpose chain =============
template <int WAVES, bool NEED_GH>
__global__ void k_edge_bwd(const float* __restrict__ rad, const float* __restrict__ drad,
                           const float* __restrict__ h,
                           const int* __restrict__ sender, const int* __restrict__ receiver,
                           const __bf16* __restrict__ R1t, const __bf16* __restrict__ R2t,
                           const __bf16* __restrict__ R3t, const __bf16* __restrict__ R4t,
                           const __bf16* __restrict__ R1row, const __bf16* __restrict__ R2row,
                           const __bf16* __restrict__ R3row, const __bf16* __restrict__ R4row,
                           const float* __restrict__ gA0, float* __restrict__ g_h,
                           float* __restrict__ g_r, int nEdges) {
  __shared__ __align__(16) __bf16 sA[WAVES][1024], sB[WAVES][1024];
  __shared__ __align__(16) __bf16 sP1[WAVES][1024], sP2[WAVES][1024], sP3[WAVES][1024];
  __shared__ float  sGR[WAVES][16 * 16];
  int wv = (int)(threadIdx.x >> 5);
  int tile = blockIdx.x * WAVES + wv;
  if (tile * 16 >= nEdges) return;
  int e0 = tile * 16;
  __bf16 *xa = sA[wv], *xb = sB[wv], *p1 = sP1[wv], *p2 = sP2[wv], *p3 = sP3[wv];
  float* gr = sGR[wv];
  int l = laneid(), lc = l & 15, lh = l >> 4;
  if (l == 0) { __builtin_prefetch(R2t, 0, 2); __builtin_prefetch(R4row, 0, 2); }

  // ---- forward recompute (save bf16 pre-activations) ----
  for (int idx = l; idx < 16 * 32; idx += 32) {
    int m = idx >> 5, k = idx & 31;
    xa[m * 64 + k] = (__bf16)((k < NBAS) ? rad[(e0 + m) * NBAS + k] : 0.f);
  }
  v8f c[4];
  {
    v16bf a = fragA(xa, 64, 0);
#pragma unroll
    for (int nt = 0; nt < 4; ++nt) {
      v8f acc = vzero8();
      acc = WMMA_BF16(a, fragB(R1t + nt * 16 * 32, 32, 0), acc);
      c[nt] = acc;
    }
  }
  storeAct<true, true>(xb, p1, c);
  gemm64(xb, R2t, c); storeAct<true, true>(xa, p2, c);
  gemm64(xa, R3t, c); storeAct<true, true>(xb, p3, c);
  gemm64(xb, R4t, c);                    // rw

  // ---- g_rw = gA0[recv] * h[send]; g_h[send] += gA0[recv] * rw ----
#pragma unroll
  for (int g = 0; g < 8; ++g) {
    int m = g + 8 * lh;
    int e = e0 + m;
    int s = sender[e], r = receiver[e];
#pragma unroll
    for (int nt = 0; nt < 4; ++nt) {
      int k = nt * 16 + lc;
      float gm = gA0[r * 64 + k];
      if (NEED_GH) atomicAdd(&g_h[s * 64 + k], gm * c[nt][g]);
      xa[m * 64 + k] = (__bf16)(gm * h[s * 64 + k]);
    }
  }
  // ---- transposed MLP chain ----
  gemm64(xa, R4row, c); multStore(xb, p3, c);
  gemm64(xb, R3row, c); multStore(xa, p2, c);
  gemm64(xa, R2row, c); multStore(xb, p1, c);
  { // g_rad (only first N-tile; cols >= 8 are zero rows of R1row)
    v16bf a0 = fragA(xb, 64, 0), a1 = fragA(xb, 64, 32);
    v8f acc = vzero8();
    acc = WMMA_BF16(a0, fragB(R1row, 64, 0),  acc);
    acc = WMMA_BF16(a1, fragB(R1row, 64, 32), acc);
#pragma unroll
    for (int g = 0; g < 8; ++g) gr[(g + 8 * lh) * 16 + lc] = acc[g];
  }
  if (l < 16) {
    int e = e0 + l;
    float s = 0.f;
    for (int b = 0; b < NBAS; ++b) s += gr[l * 16 + b] * drad[e * NBAS + b];
    g_r[e] += s;   // unique edge per lane per launch
  }
}

// ============================ node element-wise ===============================
__global__ void k_sc(const float* __restrict__ feats, const float* __restrict__ Wsc,
                     const int* __restrict__ elem, float* __restrict__ sc, int total) {
  int i = blockIdx.x * blockDim.x + threadIdx.x;
  if (i >= total) return;
  int n = i >> 6, j = i & 63;
  const float* W = Wsc + (size_t)elem[n] * KK * KK;
  float acc = 0.f;
  for (int k = 0; k < KK; ++k) acc += feats[n * KK + k] * W[k * KK + j];
  sc[i] = acc;
}

__global__ void k_poly_b0(const float* __restrict__ Aout0, const float* __restrict__ thetaL,
                          const int* __restrict__ elem, float* __restrict__ B0, int total) {
  int i = blockIdx.x * blockDim.x + threadIdx.x;
  if (i >= total) return;
  int n = i >> 6, k = i & 63;
  int z = elem[n];
  float w0 = thetaL[(0 * ZZ + z) * KK + k];
  float w1 = thetaL[(1 * ZZ + z) * KK + k];
  float w2 = thetaL[(2 * ZZ + z) * KK + k];
  float a0 = Aout0[i];
  B0[i] = (w0 + w1 * a0 + w2 * a0 * a0) * a0;     // B_pre[:, :, 0]
}

__global__ void k_node_bwd_elem(const float* __restrict__ gBpre0, const float* __restrict__ Aout0,
                                const float* __restrict__ thetaL, const int* __restrict__ elem,
                                float* __restrict__ gAout0, int total) {
  int i = blockIdx.x * blockDim.x + threadIdx.x;
  if (i >= total) return;
  int n = i >> 6, k = i & 63;
  int z = elem[n];
  float w0 = thetaL[(0 * ZZ + z) * KK + k];
  float w1 = thetaL[(1 * ZZ + z) * KK + k];
  float w2 = thetaL[(2 * ZZ + z) * KK + k];
  float a0 = Aout0[i];
  gAout0[i] = gBpre0[i] * (w0 + 2.f * w1 * a0 + 3.f * w2 * a0 * a0);
}

// ================================ readouts ====================================
__global__ void k_readout0(const float* __restrict__ feats, const float* __restrict__ wr0,
                           const int* __restrict__ batch, float* __restrict__ outE, int nN) {
  int n = blockIdx.x * blockDim.x + threadIdx.x;
  if (n >= nN) return;
  float acc = 0.f;
  for (int k = 0; k < KK; ++k) acc += feats[n * KK + k] * wr0[k];
  atomicAdd(&outE[batch[n]], acc);
}

__global__ void k_readout1(const float* __restrict__ feats, const float* __restrict__ Wr1,
                           const float* __restrict__ Wr2, const int* __restrict__ batch,
                           float* __restrict__ outE, int nN) {
  int n = blockIdx.x * blockDim.x + threadIdx.x;
  if (n >= nN) return;
  float acc = 0.f;
  for (int h = 0; h < HIDN; ++h) {
    float t = 0.f;
    for (int k = 0; k < KK; ++k) t += feats[n * KK + k] * Wr1[k * HIDN + h];
    acc += siluf(t) * Wr2[h];
  }
  atomicAdd(&outE[batch[n]], acc);
}

__global__ void k_gfeats2(const float* __restrict__ feats2, const float* __restrict__ Wr1,
                          const float* __restrict__ Wr2, float* __restrict__ gf2, int nN) {
  int n = blockIdx.x * blockDim.x + threadIdx.x;
  if (n >= nN) return;
  float gh[HIDN];
  for (int h = 0; h < HIDN; ++h) {
    float t = 0.f;
    for (int k = 0; k < KK; ++k) t += feats2[n * KK + k] * Wr1[k * HIDN + h];
    gh[h] = Wr2[h] * dsiluf(t);
  }
  for (int k = 0; k < KK; ++k) {
    float g = 0.f;
    for (int h = 0; h < HIDN; ++h) g += gh[h] * Wr1[k * HIDN + h];
    gf2[n * KK + k] = g;
  }
}

// g_feats1 = w_read0 (layer-0 readout path) + W_sc[1] path from g_feats2
__global__ void k_gsc_wread(const float* __restrict__ gf2, const float* __restrict__ Wsc1,
                            const float* __restrict__ wr0, const int* __restrict__ elem,
                            float* __restrict__ gf1, int total) {
  int i = blockIdx.x * blockDim.x + threadIdx.x;
  if (i >= total) return;
  int n = i >> 6, k = i & 63;
  const float* W = Wsc1 + (size_t)elem[n] * KK * KK + (size_t)k * KK;
  float acc = wr0[k];
  for (int j = 0; j < KK; ++j) acc += gf2[n * KK + j] * W[j];
  gf1[i] = acc;
}

// ================================ forces ======================================
__global__ void k_forces(const float* __restrict__ g_r, const float* __restrict__ u,
                         const int* __restrict__ sender, const int* __restrict__ receiver,
                         float* __restrict__ F, int nE) {
  int e = blockIdx.x * blockDim.x + threadIdx.x;
  if (e >= nE) return;
  float g = g_r[e];
  int s = sender[e], r = receiver[e];
#pragma unroll
  for (int d = 0; d < 3; ++d) {
    float v = g * u[e * 3 + d];
    atomicAdd(&F[s * 3 + d],  v);   // forces = -dE/dpos; vec = pos[r]-pos[s]
    atomicAdd(&F[r * 3 + d], -v);
  }
}

// =============================================================================
extern "C" void kernel_launch(void* const* d_in, const int* in_sizes, int n_in,
                              void* d_out, int out_size, void* d_ws, size_t ws_size,
                              hipStream_t stream) {
  (void)in_sizes; (void)n_in; (void)out_size; (void)ws_size;
  const float* positions = (const float*)d_in[0];
  const float* attrs     = (const float*)d_in[1];
  const float* shifts    = (const float*)d_in[2];
  const float* aE        = (const float*)d_in[3];
  const float* W_embed   = (const float*)d_in[4];
  const float* W_up      = (const float*)d_in[5];
  const float* R1        = (const float*)d_in[6];
  const float* R2        = (const float*)d_in[7];
  const float* R3        = (const float*)d_in[8];
  const float* R4        = (const float*)d_in[9];
  const float* W_out     = (const float*)d_in[10];
  const float* W_sc      = (const float*)d_in[11];
  const float* theta     = (const float*)d_in[12];
  const float* W_mix     = (const float*)d_in[13];
  const float* w_read0   = (const float*)d_in[14];
  const float* Wr1       = (const float*)d_in[15];
  const float* Wr2       = (const float*)d_in[16];
  const int*   edge_idx  = (const int*)d_in[17];
  const int*   batch     = (const int*)d_in[18];
  const int* sender   = edge_idx;
  const int* receiver = edge_idx + EE;

  float* outE = (float*)d_out;           // [0..G)
  float* outF = outE + GG;               // [G .. G+3N)

  // ---- workspace carve-up ----
  char* w = (char*)d_ws; size_t off = 0;
  auto alloc = [&](size_t bytes) -> void* {
    void* p = w + off; off = (off + bytes + 255) & ~((size_t)255); return p; };
  const size_t NB64 = (size_t)NN * KK * sizeof(float);
  float* u_     = (float*)alloc((size_t)EE * 3 * sizeof(float));
  float* rad    = (float*)alloc((size_t)EE * NBAS * sizeof(float));
  float* drad   = (float*)alloc((size_t)EE * NBAS * sizeof(float));
  float* g_r    = (float*)alloc((size_t)EE * sizeof(float));
  int*   elem   = (int*)  alloc((size_t)NN * sizeof(int));
  float* feats0 = (float*)alloc(NB64);
  float* feats1 = (float*)alloc(NB64);
  float* feats2 = (float*)alloc(NB64);
  float* h0     = (float*)alloc(NB64);
  float* h1     = (float*)alloc(NB64);
  float* A00    = (float*)alloc(NB64);
  float* A01    = (float*)alloc(NB64);
  float* Ao0    = (float*)alloc(NB64);
  float* Ao1    = (float*)alloc(NB64);
  float* B0     = (float*)alloc(NB64);
  float* scb    = (float*)alloc(NB64);
  float* gf2    = (float*)alloc(NB64);
  float* gf1    = (float*)alloc(NB64);
  float* gBp    = (float*)alloc(NB64);
  float* gAo    = (float*)alloc(NB64);
  float* gA0    = (float*)alloc(NB64);
  float* gh1    = (float*)alloc(NB64);
  __bf16 *R1t[2], *R1r[2], *R2t[2], *R2r[2], *R3t[2], *R3r[2], *R4t[2], *R4r[2];
  __bf16 *WupT[2], *Wupr[2], *WmxT[2], *Wmxr[2], *WotT[2], *Wotr[2];
  for (int i = 0; i < 2; ++i) {
    R1t[i]  = (__bf16*)alloc(64 * 32 * 2);  R1r[i]  = (__bf16*)alloc(16 * 64 * 2);
    R2t[i]  = (__bf16*)alloc(4096 * 2);     R2r[i]  = (__bf16*)alloc(4096 * 2);
    R3t[i]  = (__bf16*)alloc(4096 * 2);     R3r[i]  = (__bf16*)alloc(4096 * 2);
    R4t[i]  = (__bf16*)alloc(4096 * 2);     R4r[i]  = (__bf16*)alloc(4096 * 2);
    WupT[i] = (__bf16*)alloc(4096 * 2);     Wupr[i] = (__bf16*)alloc(4096 * 2);
    WmxT[i] = (__bf16*)alloc(4096 * 2);     Wmxr[i] = (__bf16*)alloc(4096 * 2);
    WotT[i] = (__bf16*)alloc(4096 * 2);     Wotr[i] = (__bf16*)alloc(4096 * 2);
  }

  // ---- per-call zeroing (atomic accumulation targets) ----
  hipMemsetAsync(d_out, 0, (size_t)(GG + NN * 3) * sizeof(float), stream);
  hipMemsetAsync(A00, 0, NB64, stream);
  hipMemsetAsync(A01, 0, NB64, stream);
  hipMemsetAsync(gh1, 0, NB64, stream);
  hipMemsetAsync(g_r, 0, (size_t)EE * sizeof(float), stream);

  auto cdiv = [](int a, int b) { return (a + b - 1) / b; };

  // ---- weight staging (bf16 column & row stages) ----
  for (int i = 0; i < 2; ++i) {
    k_stage_w<<<cdiv(2048, 256), 256, 0, stream>>>(R1 + i * NBAS * KK, NBAS, KK, 32, 16, R1t[i], R1r[i]);
    k_stage_w<<<cdiv(4096, 256), 256, 0, stream>>>(R2 + i * 4096, 64, 64, 64, 64, R2t[i], R2r[i]);
    k_stage_w<<<cdiv(4096, 256), 256, 0, stream>>>(R3 + i * 4096, 64, 64, 64, 64, R3t[i], R3r[i]);
    k_stage_w<<<cdiv(4096, 256), 256, 0, stream>>>(R4 + i * 4096, 64, 64, 64, 64, R4t[i], R4r[i]);
    k_stage_w<<<cdiv(4096, 256), 256, 0, stream>>>(W_up  + i * 4096, 64, 64, 64, 64, WupT[i], Wupr[i]);
    k_stage_w<<<cdiv(4096, 256), 256, 0, stream>>>(W_mix + i * 4096, 64, 64, 64, 64, WmxT[i], Wmxr[i]);
    k_stage_w<<<cdiv(4096, 256), 256, 0, stream>>>(W_out + (size_t)(i * 3) * 4096, 64, 64, 64, 64, WotT[i], Wotr[i]);
  }

  // ---- forward ----
  k_edge_geom<<<cdiv(EE, 256), 256, 0, stream>>>(positions, shifts, sender, receiver, u_, rad, drad, EE);
  k_elem<<<cdiv(NN, 256), 256, 0, stream>>>(attrs, aE, batch, elem, outE, NN);
  k_embed<<<cdiv(NN * KK, 256), 256, 0, stream>>>(W_embed, elem, feats0, NN * KK);

  const float* fin[2]  = {feats0, feats1};
  float*       fout[2] = {feats1, feats2};
  float*       hb[2]   = {h0, h1};
  float*       A0b[2]  = {A00, A01};
  float*       Aob[2]  = {Ao0, Ao1};
  const int gemmGrid = cdiv(cdiv(NN, 16), 8);

  for (int i = 0; i < 2; ++i) {
    const float* thetaL = theta + (size_t)(i * 3) * ZZ * KK;
    const float* WscL   = W_sc + (size_t)i * ZZ * KK * KK;
    k_gemm<8, false><<<gemmGrid, 256, 0, stream>>>(fin[i], WupT[i], nullptr, hb[i], NN, 1.f);
    k_edge_fwd<8><<<cdiv(EE / 16, 8), 256, 0, stream>>>(rad, hb[i], sender, receiver,
                                                        R1t[i], R2t[i], R3t[i], R4t[i], A0b[i], EE);
    k_gemm<8, false><<<gemmGrid, 256, 0, stream>>>(A0b[i], WotT[i], nullptr, Aob[i], NN, INV_AVG);
    k_sc<<<cdiv(NN * KK, 256), 256, 0, stream>>>(fin[i], WscL, elem, scb, NN * KK);
    k_poly_b0<<<cdiv(NN * KK, 256), 256, 0, stream>>>(Aob[i], thetaL, elem, B0, NN * KK);
    k_gemm<8, true><<<gemmGrid, 256, 0, stream>>>(B0, WmxT[i], scb, fout[i], NN, 1.f);
    if (i == 0) k_readout0<<<cdiv(NN, 256), 256, 0, stream>>>(feats1, w_read0, batch, outE, NN);
    else        k_readout1<<<cdiv(NN, 256), 256, 0, stream>>>(feats2, Wr1, Wr2, batch, outE, NN);
  }

  // ---- backward ----
  k_gfeats2<<<cdiv(NN, 256), 256, 0, stream>>>(feats2, Wr1, Wr2, gf2, NN);
  // layer 1
  {
    const float* thetaL = theta + (size_t)3 * ZZ * KK;
    const float* WscL   = W_sc + (size_t)ZZ * KK * KK;
    k_gemm<8, false><<<gemmGrid, 256, 0, stream>>>(gf2, Wmxr[1], nullptr, gBp, NN, 1.f);
    k_node_bwd_elem<<<cdiv(NN * KK, 256), 256, 0, stream>>>(gBp, Ao1, thetaL, elem, gAo, NN * KK);
    k_gemm<8, false><<<gemmGrid, 256, 0, stream>>>(gAo, Wotr[1], nullptr, gA0, NN, INV_AVG);
    k_gsc_wread<<<cdiv(NN * KK, 256), 256, 0, stream>>>(gf2, WscL, w_read0, elem, gf1, NN * KK);
    k_edge_bwd<4, true><<<cdiv(EE / 16, 4), 128, 0, stream>>>(rad, drad, h1, sender, receiver,
        R1t[1], R2t[1], R3t[1], R4t[1], R1r[1], R2r[1], R3r[1], R4r[1], gA0, gh1, g_r, EE);
    k_gemm<8, true><<<gemmGrid, 256, 0, stream>>>(gh1, Wupr[1], gf1, gf1, NN, 1.f);
  }
  // layer 0 (feats0 has no position dependence -> only the g_r path matters)
  {
    const float* thetaL = theta;
    k_gemm<8, false><<<gemmGrid, 256, 0, stream>>>(gf1, Wmxr[0], nullptr, gBp, NN, 1.f);
    k_node_bwd_elem<<<cdiv(NN * KK, 256), 256, 0, stream>>>(gBp, Ao0, thetaL, elem, gAo, NN * KK);
    k_gemm<8, false><<<gemmGrid, 256, 0, stream>>>(gAo, Wotr[0], nullptr, gA0, NN, INV_AVG);
    k_edge_bwd<4, false><<<cdiv(EE / 16, 4), 128, 0, stream>>>(rad, drad, h0, sender, receiver,
        R1t[0], R2t[0], R3t[0], R4t[0], R1r[0], R2r[0], R3r[0], R4r[0], gA0, gh1, g_r, EE);
  }
  k_forces<<<cdiv(EE, 256), 256, 0, stream>>>(g_r, u_, sender, receiver, outF, EE);
}
